// acLSTM_36017595744469
// MI455X (gfx1250) — compile-verified, matrix operand order checked by reading, not run on hardware
//
#include <hip/hip_runtime.h>
#include <hip/hip_bf16.h>

// ---------------------------------------------------------------------------
// 3-layer LSTM (H=1024) + decoder, B=128, T=200, F=171, scheduled sampling.
// MI455X / gfx1250 (wave32) strategy:
//  * Per-launch setup: weights fp32 -> bf16 pre-packed into the exact per-lane
//    V_WMMA_F32_16X16X32_BF16 B-fragment layout (32 contiguous bytes per lane
//    per 16x32 K-tile); real_seq pre-converted to padded bf16 [T][B][192].
//  * Activations bf16 row-major; A-fragment = two 16B loads per lane.
//  * Each 1-wave block computes one 16x16 tile of H for all 4 gates (shared A,
//    4 WMMAs per K-tile), f32 accumulation, LSTM cell epilogue in-register.
//  * h double-buffered per layer (cross-block recurrent RAW); c in-place
//    (each c element owned by exactly one block).
//  * Scan = 4 kernels/step (3 layers + decoder), 800 graph nodes total;
//    bf16 weights (~42 MB) stay resident in the 192 MB L2 across the scan.
// ---------------------------------------------------------------------------

typedef __attribute__((ext_vector_type(16))) __bf16 v16bf;
typedef __attribute__((ext_vector_type(8)))  __bf16 v8bf;
typedef __attribute__((ext_vector_type(8)))  float  v8f;

#define B_   128
#define T_   200
#define F_   171
#define H_   1024
#define FPAD 192      // layer-1 input K padded to a multiple of 32
#define NPD  176      // decoder N padded to a multiple of 16

// ---------------------------------------------------------------------------
// Weight pre-pack: source W is [N,K] row-major (PyTorch W, used as x @ W^T).
// Packed tile (nt,kt) of W^T occupies 512 bf16:
//   lane l (0..31), elem e (0..15):  n = nt*16 + (l&15)
//   khalf = l>>4; k = kt*32 + (e<8 ? khalf*8+e : 16 + khalf*8 + (e-8))
// Matches the 16-bit B-matrix (32x16) VGPR layout -> one contiguous 32-byte
// v16bf load per lane in the GEMM kernels.
// ---------------------------------------------------------------------------
__global__ void pack_weight_kernel(const float* __restrict__ W, __bf16* __restrict__ dst,
                                   int K, int Kpad, int N, int Npad, long total) {
  long idx = (long)blockIdx.x * blockDim.x + threadIdx.x;
  if (idx >= total) return;
  int  Ktiles = Kpad >> 5;
  int  e    = (int)(idx & 15);
  int  lane = (int)((idx >> 4) & 31);
  long tile = idx >> 9;
  int  kt   = (int)(tile % Ktiles);
  int  nt   = (int)(tile / Ktiles);
  int  n    = nt * 16 + (lane & 15);
  int  khalf = lane >> 4;
  int  k    = kt * 32 + ((e < 8) ? (khalf * 8 + e) : (16 + khalf * 8 + (e - 8)));
  float v = (k < K && n < N) ? W[(long)n * K + k] : 0.0f;
  dst[idx] = (__bf16)v;
}

__global__ void bias_sum_kernel(const float* __restrict__ a, const float* __restrict__ b,
                                float* __restrict__ o, int n) {
  int i = blockIdx.x * blockDim.x + threadIdx.x;
  if (i < n) o[i] = a[i] + b[i];
}

__global__ void bdec_pad_kernel(const float* __restrict__ b, float* __restrict__ o) {
  int i = blockIdx.x * blockDim.x + threadIdx.x;
  if (i < NPD) o[i] = (i < F_) ? b[i] : 0.0f;
}

__global__ void zero_kernel(unsigned* __restrict__ p, long n) {
  long i = (long)blockIdx.x * blockDim.x + threadIdx.x;
  if (i < n) p[i] = 0u;
}

// One-time conversion of real_seq [B,T,F] fp32 -> [T][B][FPAD] bf16 (zero pad)
__global__ void cvt_seq_kernel(const float* __restrict__ x, __bf16* __restrict__ xc) {
  long i = (long)blockIdx.x * blockDim.x + threadIdx.x;
  if (i >= (long)T_ * B_ * FPAD) return;
  int  f  = (int)(i % FPAD);
  long bt = i / FPAD;
  int  b  = (int)(bt % B_);
  int  t  = (int)(bt / B_);
  float v = (f < F_) ? x[((long)b * T_ + t) * F_ + f] : 0.0f;
  xc[i] = (__bf16)v;
}

// ---------------------------------------------------------------------------
// Fragment loaders (wave32). A: 16-bit 16x32 MxK layout; B: packed (above).
// ---------------------------------------------------------------------------
__device__ __forceinline__ v16bf load_a_frag(const __bf16* __restrict__ base,
                                             int stride, int kt) {
  const int lane = threadIdx.x & 31;
  const __bf16* p = base + (long)(lane & 15) * stride + kt * 32 + (lane >> 4) * 8;
  v8bf lo = *(const v8bf*)p;        // K = kt*32 + khalf*8 .. +7
  v8bf hi = *(const v8bf*)(p + 16); // K = kt*32 + 16 + khalf*8 .. +7
  v16bf a;
#pragma unroll
  for (int i = 0; i < 8; ++i) { a[i] = lo[i]; a[i + 8] = hi[i]; }
  return a;
}

__device__ __forceinline__ v16bf load_b_frag(const __bf16* __restrict__ packed,
                                             int Ktiles, int gnt, int kt) {
  const int lane = threadIdx.x & 31;
  return *(const v16bf*)(packed + ((long)gnt * Ktiles + kt) * 512 + lane * 16);
}

__device__ __forceinline__ float sigmf(float x) { return 1.0f / (1.0f + __expf(-x)); }

#define WMMA_BF16(acc, a, b) \
  (acc) = __builtin_amdgcn_wmma_f32_16x16x32_bf16(false, (a), false, (b), (short)0, (acc), false, false)

// ---------------------------------------------------------------------------
// One LSTM layer step: gates = actA @ W_ih^T + actH @ W_hh^T + bsum;
// cell update; writes h (bf16, double-buffered) and c (fp32, in place).
// grid = (64 n-tiles of H, 8 m-tiles of B), block = 32 (one wave).
// ---------------------------------------------------------------------------
__global__ __launch_bounds__(32) void lstm_layer_kernel(
    const __bf16* __restrict__ actA, int strideA, int KtilesA,
    const __bf16* __restrict__ pWA,
    const __bf16* __restrict__ actH,
    const __bf16* __restrict__ pWH,
    const float*  __restrict__ bsum,
    float*        __restrict__ c,
    __bf16*       __restrict__ hout) {
  const int nt = blockIdx.x;  // column tile within H
  const int mt = blockIdx.y;  // row tile within B
  v8f a0 = {}, a1 = {}, a2 = {}, a3 = {};

  const __bf16* aBase = actA + (long)mt * 16 * strideA;
#pragma unroll 2
  for (int kt = 0; kt < KtilesA; ++kt) {
    v16bf a  = load_a_frag(aBase, strideA, kt);
    v16bf b0 = load_b_frag(pWA, KtilesA, 0 * 64 + nt, kt);
    v16bf b1 = load_b_frag(pWA, KtilesA, 1 * 64 + nt, kt);
    v16bf b2 = load_b_frag(pWA, KtilesA, 2 * 64 + nt, kt);
    v16bf b3 = load_b_frag(pWA, KtilesA, 3 * 64 + nt, kt);
    WMMA_BF16(a0, a, b0); WMMA_BF16(a1, a, b1);
    WMMA_BF16(a2, a, b2); WMMA_BF16(a3, a, b3);
  }
  const __bf16* hBase = actH + (long)mt * 16 * H_;
#pragma unroll 2
  for (int kt = 0; kt < (H_ >> 5); ++kt) {
    v16bf a  = load_a_frag(hBase, H_, kt);
    v16bf b0 = load_b_frag(pWH, H_ >> 5, 0 * 64 + nt, kt);
    v16bf b1 = load_b_frag(pWH, H_ >> 5, 1 * 64 + nt, kt);
    v16bf b2 = load_b_frag(pWH, H_ >> 5, 2 * 64 + nt, kt);
    v16bf b3 = load_b_frag(pWH, H_ >> 5, 3 * 64 + nt, kt);
    WMMA_BF16(a0, a, b0); WMMA_BF16(a1, a, b1);
    WMMA_BF16(a2, a, b2); WMMA_BF16(a3, a, b3);
  }

  // Epilogue. C/D layout: VGPR r, lane l -> M = (l>=16)*8 + r, N = l&15.
  const int lane = threadIdx.x & 31;
  const int n  = nt * 16 + (lane & 15);
  const int m0 = mt * 16 + ((lane >> 4) << 3);
  const float bi = bsum[n],          bfv = bsum[H_ + n];
  const float bg = bsum[2 * H_ + n], bo  = bsum[3 * H_ + n];
#pragma unroll
  for (int r = 0; r < 8; ++r) {
    long idx = (long)(m0 + r) * H_ + n;
    float gi = a0[r] + bi, gf = a1[r] + bfv, gg = a2[r] + bg, go = a3[r] + bo;
    float cn = sigmf(gf) * c[idx] + sigmf(gi) * tanhf(gg);
    c[idx]   = cn;
    hout[idx] = (__bf16)(sigmf(go) * tanhf(cn));
  }
}

// ---------------------------------------------------------------------------
// Decoder: out = h2 @ W_dec^T + b_dec. Writes d_out[:, t*F:(t+1)*F] (fp32)
// and prev_out (bf16, row stride FPAD; cols 171..175 get exact zeros via
// zero-padded weights/bias, cols 176..191 zeroed at init and never touched).
// grid = (11 n-tiles of NPD, 8 m-tiles), block = 32.
// ---------------------------------------------------------------------------
__global__ __launch_bounds__(32) void decoder_kernel(
    const __bf16* __restrict__ h,
    const __bf16* __restrict__ pWdec,
    const float*  __restrict__ bdec,
    float*        __restrict__ out,
    __bf16*       __restrict__ prev_out, int t) {
  const int nt = blockIdx.x, mt = blockIdx.y;
  v8f acc = {};
  const __bf16* hb = h + (long)mt * 16 * H_;
#pragma unroll 2
  for (int kt = 0; kt < (H_ >> 5); ++kt) {
    v16bf a = load_a_frag(hb, H_, kt);
    v16bf b = load_b_frag(pWdec, H_ >> 5, nt, kt);
    WMMA_BF16(acc, a, b);
  }
  const int lane = threadIdx.x & 31;
  const int n  = nt * 16 + (lane & 15);
  const int m0 = mt * 16 + ((lane >> 4) << 3);
  const float bias = bdec[n];
#pragma unroll
  for (int r = 0; r < 8; ++r) {
    int m = m0 + r;
    float v = acc[r] + bias;
    prev_out[m * FPAD + n] = (__bf16)v;
    if (n < F_) out[(long)m * (T_ * F_) + (long)t * F_ + n] = v;
  }
}

// ---------------------------------------------------------------------------
extern "C" void kernel_launch(void* const* d_in, const int* in_sizes, int n_in,
                              void* d_out, int out_size, void* d_ws, size_t ws_size,
                              hipStream_t stream) {
  const float* x     = (const float*)d_in[0];
  const float* Wih1  = (const float*)d_in[1];  const float* bih1 = (const float*)d_in[2];
  const float* Whh1  = (const float*)d_in[3];  const float* bhh1 = (const float*)d_in[4];
  const float* Wih2  = (const float*)d_in[5];  const float* bih2 = (const float*)d_in[6];
  const float* Whh2  = (const float*)d_in[7];  const float* bhh2 = (const float*)d_in[8];
  const float* Wih3  = (const float*)d_in[9];  const float* bih3 = (const float*)d_in[10];
  const float* Whh3  = (const float*)d_in[11]; const float* bhh3 = (const float*)d_in[12];
  const float* Wdec  = (const float*)d_in[13]; const float* bdec = (const float*)d_in[14];

  char* p = (char*)d_ws;
  auto carve = [&](size_t bytes) -> char* {
    char* r = p; p += (bytes + 255) & ~(size_t)255; return r;
  };

  // packed bf16 weights (~42 MB total; L2-resident on MI455X's 192 MB L2)
  const long szIH1 = (long)(4096 / 16) * (FPAD / 32) * 512;   // elements
  const long szHH  = (long)(4096 / 16) * (H_ / 32) * 512;
  const long szDEC = (long)(NPD / 16) * (H_ / 32) * 512;
  const long szSEQ = (long)T_ * B_ * FPAD;
  __bf16* pWih1 = (__bf16*)carve(szIH1 * 2);
  __bf16* pWhh1 = (__bf16*)carve(szHH * 2);
  __bf16* pWih2 = (__bf16*)carve(szHH * 2);
  __bf16* pWhh2 = (__bf16*)carve(szHH * 2);
  __bf16* pWih3 = (__bf16*)carve(szHH * 2);
  __bf16* pWhh3 = (__bf16*)carve(szHH * 2);
  __bf16* pWdec = (__bf16*)carve(szDEC * 2);
  __bf16* xcvt  = (__bf16*)carve(szSEQ * 2);  // [T][B][FPAD]
  float* bsum1 = (float*)carve(4096 * 4);
  float* bsum2 = (float*)carve(4096 * 4);
  float* bsum3 = (float*)carve(4096 * 4);
  float* bdecp = (float*)carve(NPD * 4);

  // state (zeroed every call for determinism)
  char* state0 = p;
  __bf16* h0a = (__bf16*)carve(B_ * H_ * 2);
  __bf16* h0b = (__bf16*)carve(B_ * H_ * 2);
  __bf16* h1a = (__bf16*)carve(B_ * H_ * 2);
  __bf16* h1b = (__bf16*)carve(B_ * H_ * 2);
  __bf16* h2a = (__bf16*)carve(B_ * H_ * 2);
  __bf16* h2b = (__bf16*)carve(B_ * H_ * 2);
  float*  c0  = (float*)carve(B_ * H_ * 4);
  float*  c1  = (float*)carve(B_ * H_ * 4);
  float*  c2  = (float*)carve(B_ * H_ * 4);
  __bf16* pout = (__bf16*)carve(B_ * FPAD * 2);
  long state_dw = (long)(p - state0) / 4;

  // --- one-time (per launch) prep, all on `stream` ---
  zero_kernel<<<dim3((unsigned)((state_dw + 255) / 256)), 256, 0, stream>>>((unsigned*)state0, state_dw);

  pack_weight_kernel<<<dim3((unsigned)((szIH1 + 255) / 256)), 256, 0, stream>>>(Wih1, pWih1, F_, FPAD, 4096, 4096, szIH1);
  pack_weight_kernel<<<dim3((unsigned)((szHH + 255) / 256)), 256, 0, stream>>>(Whh1, pWhh1, H_, H_, 4096, 4096, szHH);
  pack_weight_kernel<<<dim3((unsigned)((szHH + 255) / 256)), 256, 0, stream>>>(Wih2, pWih2, H_, H_, 4096, 4096, szHH);
  pack_weight_kernel<<<dim3((unsigned)((szHH + 255) / 256)), 256, 0, stream>>>(Whh2, pWhh2, H_, H_, 4096, 4096, szHH);
  pack_weight_kernel<<<dim3((unsigned)((szHH + 255) / 256)), 256, 0, stream>>>(Wih3, pWih3, H_, H_, 4096, 4096, szHH);
  pack_weight_kernel<<<dim3((unsigned)((szHH + 255) / 256)), 256, 0, stream>>>(Whh3, pWhh3, H_, H_, 4096, 4096, szHH);
  pack_weight_kernel<<<dim3((unsigned)((szDEC + 255) / 256)), 256, 0, stream>>>(Wdec, pWdec, H_, H_, F_, NPD, szDEC);
  cvt_seq_kernel<<<dim3((unsigned)((szSEQ + 255) / 256)), 256, 0, stream>>>(x, xcvt);

  bias_sum_kernel<<<dim3(16), 256, 0, stream>>>(bih1, bhh1, bsum1, 4096);
  bias_sum_kernel<<<dim3(16), 256, 0, stream>>>(bih2, bhh2, bsum2, 4096);
  bias_sum_kernel<<<dim3(16), 256, 0, stream>>>(bih3, bhh3, bsum3, 4096);
  bdec_pad_kernel<<<dim3(1), 256, 0, stream>>>(bdec, bdecp);

  // --- serial scan over T: 4 kernels/step, all ordered on `stream` ---
  __bf16 *h0i = h0a, *h0o = h0b, *h1i = h1a, *h1o = h1b, *h2i = h2a, *h2o = h2b;
  const dim3 gridL(64, 8), gridD(NPD / 16, 8);
  for (int t = 0; t < T_; ++t) {
    int use_gt = (t % 10) < 5;  // 5 ground-truth steps, 5 autoregressive
    const __bf16* a1 = use_gt ? (xcvt + (long)t * B_ * FPAD) : pout;
    lstm_layer_kernel<<<gridL, 32, 0, stream>>>(a1,  FPAD, FPAD / 32, pWih1, h0i, pWhh1, bsum1, c0, h0o);
    lstm_layer_kernel<<<gridL, 32, 0, stream>>>(h0o, H_,   H_ / 32,   pWih2, h1i, pWhh2, bsum2, c1, h1o);
    lstm_layer_kernel<<<gridL, 32, 0, stream>>>(h1o, H_,   H_ / 32,   pWih3, h2i, pWhh3, bsum3, c2, h2o);
    decoder_kernel<<<gridD, 32, 0, stream>>>(h2o, pWdec, bdecp, (float*)d_out, pout, t);
    __bf16* tsw;
    tsw = h0i; h0i = h0o; h0o = tsw;
    tsw = h1i; h1i = h1o; h1o = tsw;
    tsw = h2i; h2i = h2o; h2o = tsw;
  }
}
